// ConvSeq2Seq_60060822667683
// MI455X (gfx1250) — compile-verified, matrix operand order checked by reading, not run on hardware
//
#include <hip/hip_runtime.h>
#include <hip/hip_bf16.h>

// ---------------------------------------------------------------------------
// ConvSeq2Seq for MI455X (gfx1250, wave32, WMMA).
// Pipeline:
//   1. k_pack_w / k_bias2 / k_pack_noise : repack LSTM weights to f16 WMMA
//      B-layout, combine biases, repack decoder noise to WMMA A-layout.
//   2. k_summarize : conv stack (1050 -> 64ch frame), writes packed A-tiles.
//   3. k_lstm_scan : persistent 1-workgroup 16-wave kernel; 512 encoder +
//      512 decoder steps of 2-layer LSTM using v_wmma_f32_16x16x32_f16,
//      state in LDS, plus decoder fc GEMM producing v[t] frames.
//   4. k_desummarize : deconv stack v[t] -> 1050 samples, parallel.
// ---------------------------------------------------------------------------

#define B_    16
#define H_    512
#define CS_   64
#define S_    512
#define T_    512
#define SEG_  1050
#define L_    (T_ * SEG_)
#define G_    (4 * H_)        // 2048 gate columns
#define NT_   (G_ / 16)       // 128 N-tiles
#define KCH_H (H_ / 32)       // 16 k-chunks for H-sized K
#define KCH_X (CS_ / 32)      // 2 k-chunks for CS-sized K
#define CHUNK 512             // halfs per packed 16x32 (or 32x16) tile

typedef __attribute__((ext_vector_type(16))) _Float16 v16h;
typedef __attribute__((ext_vector_type(8)))  float    v8f;

// ---- 16-bit WMMA A/B lane/half layout (ISA 7.12.2) ------------------------
// forward map: lane l, half i -> K offset within 32-wide chunk
__device__ __forceinline__ int kk_of(int lane, int i) {
    return i + ((lane >= 16) ? 8 : 0) + ((i >= 8) ? 8 : 0);
}
// inverse map: row m (0..15), K offset kk (0..31) -> (lane, half)
__device__ __forceinline__ void packA_store(_Float16* chunk, int m, int kk, float v) {
    int lane = m + (((kk >> 3) & 1) << 4);
    int i = (kk < 8) ? kk : (kk < 16) ? kk - 8 : (kk < 24) ? kk - 8 : kk - 16;
    chunk[lane * 16 + i] = (_Float16)v;
}

__device__ __forceinline__ v8f wmma_acc(v8f acc, const _Float16* A,
                                        const _Float16* Bm, int lane) {
    v16h a = *(const v16h*)(A + lane * 16);
    v16h b = *(const v16h*)(Bm + lane * 16);
    return __builtin_amdgcn_wmma_f32_16x16x32_f16(
        /*neg_a=*/false, a, /*neg_b=*/false, b,
        /*c_mod=*/(short)0, acc, /*reuse_a=*/false, /*reuse_b=*/false);
}

__device__ __forceinline__ float sigmoid_(float x) {
    return 1.0f / (1.0f + __expf(-x));
}

// ---------------------------------------------------------------------------
// Weight repack: W[N x K] (z = x @ W^T) -> B-matrix tiles
// dst[ntile][kchunk][lane][half], element = W[ntile*16 + lane%16][kchunk*32+kk]
// ---------------------------------------------------------------------------
__global__ void k_pack_w(const float* __restrict__ W, _Float16* __restrict__ dst,
                         int N, int K) {
    int idx = blockIdx.x * blockDim.x + threadIdx.x;
    if (idx >= N * K) return;
    int n = idx / K, k = idx % K;
    int ntile = n >> 4, ncol = n & 15;
    int kch = k >> 5, kk = k & 31;
    int lane = ncol + (((kk >> 3) & 1) << 4);
    int i = (kk < 8) ? kk : (kk < 16) ? kk - 8 : (kk < 24) ? kk - 8 : kk - 16;
    int KCH = K >> 5;
    dst[(((size_t)ntile * KCH + kch) << 9) + lane * 16 + i] = (_Float16)W[idx];
}

__global__ void k_bias2(const float* __restrict__ a, const float* __restrict__ b,
                        float* __restrict__ o, int n) {
    int i = blockIdx.x * blockDim.x + threadIdx.x;
    if (i < n) o[i] = a[i] + b[i];
}

// dec_noise [T,B,64] f32 -> packed A tiles [T][2][32][16] f16
__global__ void k_pack_noise(const float* __restrict__ noise, _Float16* __restrict__ dst) {
    int idx = blockIdx.x * blockDim.x + threadIdx.x;
    if (idx >= T_ * B_ * CS_) return;
    int t = idx / (B_ * CS_);
    int r = idx % (B_ * CS_);
    int b = r / CS_, c = r % CS_;
    packA_store(dst + ((size_t)t * KCH_X + (c >> 5)) * CHUNK, b, c & 31, noise[idx]);
}

// ---------------------------------------------------------------------------
// Encoder conv stack: one block per (b,s) frame; writes packed A-tile f16.
// ---------------------------------------------------------------------------
__global__ void k_summarize(const float* __restrict__ prev,
                            const float* __restrict__ c1w, const float* __restrict__ c1b,
                            const float* __restrict__ c2w, const float* __restrict__ c2b,
                            const float* __restrict__ c3w, const float* __restrict__ c3b,
                            const float* __restrict__ c4w, const float* __restrict__ c4b,
                            _Float16* __restrict__ xE) {
    int frame = blockIdx.x;              // b*S_ + s
    int b = frame / S_, s = frame % S_;
    const float* x = prev + (size_t)b * L_ + (size_t)s * SEG_;
    __shared__ float s1[2 * 350], s2[8 * 70], s3[32 * 10];

    for (int idx = threadIdx.x; idx < 2 * 350; idx += blockDim.x) {
        int co = idx / 350, p = idx % 350;
        float a = c1b[co];
#pragma unroll
        for (int k = 0; k < 3; ++k) a += x[p * 3 + k] * c1w[co * 3 + k];
        s1[co * 350 + p] = tanhf(a);
    }
    __syncthreads();
    for (int idx = threadIdx.x; idx < 8 * 70; idx += blockDim.x) {
        int co = idx / 70, p = idx % 70;
        float a = c2b[co];
        for (int ci = 0; ci < 2; ++ci)
#pragma unroll
            for (int k = 0; k < 5; ++k)
                a += s1[ci * 350 + p * 5 + k] * c2w[(co * 2 + ci) * 5 + k];
        s2[co * 70 + p] = tanhf(a);
    }
    __syncthreads();
    for (int idx = threadIdx.x; idx < 32 * 10; idx += blockDim.x) {
        int co = idx / 10, p = idx % 10;
        float a = c3b[co];
        for (int ci = 0; ci < 8; ++ci)
#pragma unroll
            for (int k = 0; k < 7; ++k)
                a += s2[ci * 70 + p * 7 + k] * c3w[(co * 8 + ci) * 7 + k];
        s3[co * 10 + p] = tanhf(a);
    }
    __syncthreads();
    for (int co = threadIdx.x; co < 64; co += blockDim.x) {
        float a = c4b[co];
        for (int ci = 0; ci < 32; ++ci)
#pragma unroll
            for (int k = 0; k < 10; ++k)
                a += s3[ci * 10 + k] * c4w[(co * 32 + ci) * 10 + k];
        packA_store(xE + ((size_t)s * KCH_X + (co >> 5)) * CHUNK, b, co & 31, a);
    }
}

// ---------------------------------------------------------------------------
// Persistent LSTM scan: 1 workgroup, 512 threads = 16 waves, WMMA GEMMs.
// ---------------------------------------------------------------------------
__device__ __forceinline__ void lstm_update(const float* __restrict__ zbuf,
                                            float* __restrict__ cbuf,
                                            _Float16* __restrict__ hp,
                                            _Float16* __restrict__ htp, int tid) {
    for (int e = tid; e < B_ * H_; e += 512) {
        int b = e >> 9;
        int j = e & (H_ - 1);
        const float* zr = zbuf + b * G_;
        float ii = sigmoid_(zr[j]);
        float ff = sigmoid_(zr[H_ + j]);
        float gg = tanhf(zr[2 * H_ + j]);
        float oo = sigmoid_(zr[3 * H_ + j]);
        float cn = ff * cbuf[e] + ii * gg;
        float hn = oo * tanhf(cn);
        cbuf[e] = cn;
        packA_store(hp + ((size_t)(j >> 5) << 9), b, j & 31, hn);
        if (htp) packA_store(htp + ((size_t)(j >> 5) << 9), b, j & 31, tanhf(hn));
    }
}

__global__ __launch_bounds__(512) void k_lstm_scan(
    const _Float16* __restrict__ xE, const _Float16* __restrict__ xD,
    const _Float16* __restrict__ We0x, const _Float16* __restrict__ We0h,
    const _Float16* __restrict__ We1x, const _Float16* __restrict__ We1h,
    const _Float16* __restrict__ Wr0x, const _Float16* __restrict__ Wr0h,
    const _Float16* __restrict__ Wr1x, const _Float16* __restrict__ Wr1h,
    const _Float16* __restrict__ Wfc,
    const float* __restrict__ be0, const float* __restrict__ be1,
    const float* __restrict__ br0, const float* __restrict__ br1,
    const float* __restrict__ fcb,
    float* __restrict__ vout) {
    extern __shared__ char dynsmem[];
    _Float16* h0p = (_Float16*)dynsmem;                 // 16 chunks, 16 KB
    _Float16* h1p = h0p + KCH_H * CHUNK;                // 16 KB
    _Float16* htp = h1p + KCH_H * CHUNK;                // 16 KB (tanh(h1) for fc)
    float* c0 = (float*)(htp + KCH_H * CHUNK);          // 32 KB
    float* c1 = c0 + B_ * H_;                           // 32 KB
    float* zbuf = c1 + B_ * H_;                         // 16 x 2048 f32, 128 KB

    const int tid = threadIdx.x;
    const int wave = tid >> 5;
    const int lane = tid & 31;

    for (int e = tid; e < B_ * H_; e += 512) { c0[e] = 0.f; c1[e] = 0.f; }
    for (int e = tid; e < KCH_H * CHUNK; e += 512) {
        h0p[e] = (_Float16)0.f; h1p[e] = (_Float16)0.f; htp[e] = (_Float16)0.f;
    }
    __syncthreads();

    for (int s = 0; s < S_ + T_; ++s) {
        const bool enc = (s < S_);
        const _Float16* Xp = enc ? xE + (size_t)s * (KCH_X * CHUNK)
                                 : xD + (size_t)(s - S_) * (KCH_X * CHUNK);
        const _Float16* Wx0 = enc ? We0x : Wr0x;
        const _Float16* Wh0 = enc ? We0h : Wr0h;
        const _Float16* Wx1 = enc ? We1x : Wr1x;
        const _Float16* Wh1 = enc ? We1h : Wr1h;
        const float* b0 = enc ? be0 : br0;
        const float* b1 = enc ? be1 : br1;

        // ---- layer 0: z = x_t @ Wih0^T + h0 @ Whh0^T + b0 ----
        for (int t = 0; t < 8; ++t) {
            int ntile = wave * 8 + t;
            float bv = b0[ntile * 16 + (lane & 15)];
            v8f acc = {bv, bv, bv, bv, bv, bv, bv, bv};
            const _Float16* wx = Wx0 + (size_t)ntile * (KCH_X * CHUNK);
#pragma unroll
            for (int k = 0; k < KCH_X; ++k)
                acc = wmma_acc(acc, Xp + k * CHUNK, wx + k * CHUNK, lane);
            const _Float16* wh = Wh0 + (size_t)ntile * (KCH_H * CHUNK);
            if (t < 7)  // speculative stream-ahead of next tile's weights
                __builtin_prefetch(Wh0 + (size_t)(ntile + 1) * (KCH_H * CHUNK) + lane * 16, 0, 1);
#pragma unroll 4
            for (int k = 0; k < KCH_H; ++k)
                acc = wmma_acc(acc, h0p + k * CHUNK, wh + k * CHUNK, lane);
            int col = ntile * 16 + (lane & 15);
            int rb = (lane >> 4) * 8;
#pragma unroll
            for (int r = 0; r < 8; ++r) zbuf[(rb + r) * G_ + col] = acc[r];
        }
        __syncthreads();
        lstm_update(zbuf, c0, h0p, nullptr, tid);
        __syncthreads();

        // ---- layer 1: z = h0 @ Wih1^T + h1 @ Whh1^T + b1 ----
        for (int t = 0; t < 8; ++t) {
            int ntile = wave * 8 + t;
            float bv = b1[ntile * 16 + (lane & 15)];
            v8f acc = {bv, bv, bv, bv, bv, bv, bv, bv};
            const _Float16* wx = Wx1 + (size_t)ntile * (KCH_H * CHUNK);
#pragma unroll 4
            for (int k = 0; k < KCH_H; ++k)
                acc = wmma_acc(acc, h0p + k * CHUNK, wx + k * CHUNK, lane);
            const _Float16* wh = Wh1 + (size_t)ntile * (KCH_H * CHUNK);
            if (t < 7)
                __builtin_prefetch(Wh1 + (size_t)(ntile + 1) * (KCH_H * CHUNK) + lane * 16, 0, 1);
#pragma unroll 4
            for (int k = 0; k < KCH_H; ++k)
                acc = wmma_acc(acc, h1p + k * CHUNK, wh + k * CHUNK, lane);
            int col = ntile * 16 + (lane & 15);
            int rb = (lane >> 4) * 8;
#pragma unroll
            for (int r = 0; r < 8; ++r) zbuf[(rb + r) * G_ + col] = acc[r];
        }
        __syncthreads();
        lstm_update(zbuf, c1, h1p, enc ? nullptr : htp, tid);
        __syncthreads();

        // ---- decoder fc: v = tanh(h1) @ fcw^T + fcb  (16x64, 4 waves) ----
        if (!enc) {
            if (wave < 4) {
                int ntile = wave;
                float bv = fcb[ntile * 16 + (lane & 15)];
                v8f acc = {bv, bv, bv, bv, bv, bv, bv, bv};
                const _Float16* wf = Wfc + (size_t)ntile * (KCH_H * CHUNK);
#pragma unroll 4
                for (int k = 0; k < KCH_H; ++k)
                    acc = wmma_acc(acc, htp + k * CHUNK, wf + k * CHUNK, lane);
                int col = ntile * 16 + (lane & 15);
                int rb = (lane >> 4) * 8;
                int t = s - S_;
#pragma unroll
                for (int r = 0; r < 8; ++r)
                    vout[((size_t)t * B_ + rb + r) * CS_ + col] = acc[r];
            }
            __syncthreads();
        }
    }
}

// ---------------------------------------------------------------------------
// Decoder deconv stack: one block per (b,t) frame.
// ---------------------------------------------------------------------------
__global__ void k_desummarize(const float* __restrict__ vout,
                              const float* __restrict__ d1w, const float* __restrict__ d1b,
                              const float* __restrict__ d2w, const float* __restrict__ d2b,
                              const float* __restrict__ d3w, const float* __restrict__ d3b,
                              const float* __restrict__ d4w, const float* __restrict__ d4b,
                              float* __restrict__ out) {
    int frame = blockIdx.x;              // b*T_ + t
    int b = frame / T_, t = frame % T_;
    __shared__ float vv[64], s1[32 * 10], s2[8 * 70], s3[2 * 350];

    for (int c = threadIdx.x; c < 64; c += blockDim.x)
        vv[c] = vout[((size_t)t * B_ + b) * CS_ + c];
    __syncthreads();
    for (int idx = threadIdx.x; idx < 320; idx += blockDim.x) {
        int co = idx / 10, kk = idx % 10;
        float a = d1b[co];
        for (int ci = 0; ci < 64; ++ci) a += vv[ci] * d1w[ci * 320 + co * 10 + kk];
        s1[idx] = tanhf(a);
    }
    __syncthreads();
    for (int idx = threadIdx.x; idx < 560; idx += blockDim.x) {
        int co = idx / 70, r = idx % 70, xq = r / 7, kk = r % 7;
        float a = d2b[co];
        for (int ci = 0; ci < 32; ++ci) a += s1[ci * 10 + xq] * d2w[ci * 56 + co * 7 + kk];
        s2[idx] = tanhf(a);
    }
    __syncthreads();
    for (int idx = threadIdx.x; idx < 700; idx += blockDim.x) {
        int co = idx / 350, r = idx % 350, xq = r / 5, kk = r % 5;
        float a = d3b[co];
        for (int ci = 0; ci < 8; ++ci) a += s2[ci * 70 + xq] * d3w[ci * 10 + co * 5 + kk];
        s3[idx] = tanhf(a);
    }
    __syncthreads();
    float* o = out + (size_t)b * L_ + (size_t)t * SEG_;
    for (int idx = threadIdx.x; idx < SEG_; idx += blockDim.x) {
        int xq = idx / 3, kk = idx % 3;
        float a = d4b[0];
        a += s3[xq] * d4w[kk];
        a += s3[350 + xq] * d4w[3 + kk];
        o[idx] = a;
    }
}

// ---------------------------------------------------------------------------
extern "C" void kernel_launch(void* const* d_in, const int* in_sizes, int n_in,
                              void* d_out, int out_size, void* d_ws, size_t ws_size,
                              hipStream_t stream) {
    const float* prev      = (const float*)d_in[0];
    const float* dec_noise = (const float*)d_in[2];
    const float* c1w = (const float*)d_in[3];  const float* c1b = (const float*)d_in[4];
    const float* c2w = (const float*)d_in[5];  const float* c2b = (const float*)d_in[6];
    const float* c3w = (const float*)d_in[7];  const float* c3b = (const float*)d_in[8];
    const float* c4w = (const float*)d_in[9];  const float* c4b = (const float*)d_in[10];
    const float* d1w = (const float*)d_in[11]; const float* d1b = (const float*)d_in[12];
    const float* d2w = (const float*)d_in[13]; const float* d2b = (const float*)d_in[14];
    const float* d3w = (const float*)d_in[15]; const float* d3b = (const float*)d_in[16];
    const float* d4w = (const float*)d_in[17]; const float* d4b = (const float*)d_in[18];
    const float* fcw = (const float*)d_in[19]; const float* fcb = (const float*)d_in[20];
    const float* e_wih0 = (const float*)d_in[21]; const float* e_whh0 = (const float*)d_in[22];
    const float* e_bih0 = (const float*)d_in[23]; const float* e_bhh0 = (const float*)d_in[24];
    const float* e_wih1 = (const float*)d_in[25]; const float* e_whh1 = (const float*)d_in[26];
    const float* e_bih1 = (const float*)d_in[27]; const float* e_bhh1 = (const float*)d_in[28];
    const float* r_wih0 = (const float*)d_in[29]; const float* r_whh0 = (const float*)d_in[30];
    const float* r_bih0 = (const float*)d_in[31]; const float* r_bhh0 = (const float*)d_in[32];
    const float* r_wih1 = (const float*)d_in[33]; const float* r_whh1 = (const float*)d_in[34];
    const float* r_bih1 = (const float*)d_in[35]; const float* r_bhh1 = (const float*)d_in[36];
    float* out = (float*)d_out;

    // workspace carve-out (bump allocator, 256B aligned)
    char* ws = (char*)d_ws;
    auto alloc = [&](size_t bytes) -> char* {
        char* p = ws;
        ws += (bytes + 255) & ~(size_t)255;
        return p;
    };
    _Float16* We0x = (_Float16*)alloc((size_t)NT_ * KCH_X * CHUNK * 2);
    _Float16* We0h = (_Float16*)alloc((size_t)NT_ * KCH_H * CHUNK * 2);
    _Float16* We1x = (_Float16*)alloc((size_t)NT_ * KCH_H * CHUNK * 2);
    _Float16* We1h = (_Float16*)alloc((size_t)NT_ * KCH_H * CHUNK * 2);
    _Float16* Wr0x = (_Float16*)alloc((size_t)NT_ * KCH_X * CHUNK * 2);
    _Float16* Wr0h = (_Float16*)alloc((size_t)NT_ * KCH_H * CHUNK * 2);
    _Float16* Wr1x = (_Float16*)alloc((size_t)NT_ * KCH_H * CHUNK * 2);
    _Float16* Wr1h = (_Float16*)alloc((size_t)NT_ * KCH_H * CHUNK * 2);
    _Float16* Wfc  = (_Float16*)alloc((size_t)(CS_ / 16) * KCH_H * CHUNK * 2);
    _Float16* xE   = (_Float16*)alloc((size_t)S_ * KCH_X * CHUNK * 2);
    _Float16* xD   = (_Float16*)alloc((size_t)T_ * KCH_X * CHUNK * 2);
    float* be0 = (float*)alloc(G_ * 4);
    float* be1 = (float*)alloc(G_ * 4);
    float* br0 = (float*)alloc(G_ * 4);
    float* br1 = (float*)alloc(G_ * 4);
    float* vout = (float*)alloc((size_t)T_ * B_ * CS_ * 4);

    // --- stage 1: repack weights / biases / noise ---
    auto packw = [&](const float* W, _Float16* dst, int N, int K) {
        int tot = N * K;
        k_pack_w<<<(tot + 255) / 256, 256, 0, stream>>>(W, dst, N, K);
    };
    packw(e_wih0, We0x, G_, CS_);
    packw(e_whh0, We0h, G_, H_);
    packw(e_wih1, We1x, G_, H_);
    packw(e_whh1, We1h, G_, H_);
    packw(r_wih0, Wr0x, G_, CS_);
    packw(r_whh0, Wr0h, G_, H_);
    packw(r_wih1, Wr1x, G_, H_);
    packw(r_whh1, Wr1h, G_, H_);
    packw(fcw, Wfc, CS_, H_);

    k_bias2<<<(G_ + 255) / 256, 256, 0, stream>>>(e_bih0, e_bhh0, be0, G_);
    k_bias2<<<(G_ + 255) / 256, 256, 0, stream>>>(e_bih1, e_bhh1, be1, G_);
    k_bias2<<<(G_ + 255) / 256, 256, 0, stream>>>(r_bih0, r_bhh0, br0, G_);
    k_bias2<<<(G_ + 255) / 256, 256, 0, stream>>>(r_bih1, r_bhh1, br1, G_);

    k_pack_noise<<<(T_ * B_ * CS_ + 255) / 256, 256, 0, stream>>>(dec_noise, xD);

    // --- stage 2: encoder conv summarize (parallel over 8192 frames) ---
    k_summarize<<<B_ * S_, 128, 0, stream>>>(prev, c1w, c1b, c2w, c2b, c3w, c3b,
                                             c4w, c4b, xE);

    // --- stage 3: persistent WMMA LSTM scan (encoder + decoder) ---
    const size_t smem = (size_t)(3 * KCH_H * CHUNK) * 2        // h0p,h1p,htp f16
                      + (size_t)(2 * B_ * H_) * 4              // c0,c1 f32
                      + (size_t)(B_ * G_) * 4;                 // zbuf f32
    k_lstm_scan<<<1, 512, smem, stream>>>(xE, xD,
                                          We0x, We0h, We1x, We1h,
                                          Wr0x, Wr0h, Wr1x, Wr1h, Wfc,
                                          be0, be1, br0, br1, fcb, vout);

    // --- stage 4: decoder deconv desummarize (parallel over 8192 frames) ---
    k_desummarize<<<B_ * T_, 128, 0, stream>>>(vout, d1w, d1b, d2w, d2b,
                                               d3w, d3b, d4w, d4b, out);
}
